// MAGClassifier_26439818674304
// MI455X (gfx1250) — compile-verified, matrix operand order checked by reading, not run on hardware
//
#include <hip/hip_runtime.h>

// ---------------- problem constants ----------------
#define B_   64
#define N_   128
#define E_   512
#define ND_  32
#define ED_  16
#define HID_ 128
#define H_   8
#define M_   (B_ * E_)      // 32768 edge-tokens total
#define KEF_ 96             // 80 (2*ND+ED) padded to multiple of 32

typedef __attribute__((ext_vector_type(16))) _Float16 v16h;
typedef __attribute__((ext_vector_type(8)))  _Float16 v8h;
typedef __attribute__((ext_vector_type(8)))  float    v8f;
typedef int v4i __attribute__((vector_size(16)));

__device__ __forceinline__ float mishf(float x) {
  float sp = (x > 20.f) ? x : log1pf(__expf(x));
  return x * tanhf(sp);
}

// 16-byte global->LDS copy, async (ASYNCcnt) when the toolchain exposes it.
__device__ __forceinline__ void copy16_g2l(const _Float16* g, _Float16* l) {
#if __has_builtin(__builtin_amdgcn_global_load_async_to_lds_b128)
  __builtin_amdgcn_global_load_async_to_lds_b128(
      (__attribute__((address_space(1))) v4i*)g,
      (__attribute__((address_space(3))) v4i*)l, 0, 0);
#else
  *(v8h*)l = *(const v8h*)g;
#endif
}
__device__ __forceinline__ void copy_drain() {
#if __has_builtin(__builtin_amdgcn_s_wait_asynccnt)
  __builtin_amdgcn_s_wait_asynccnt(0);
#elif __has_builtin(__builtin_amdgcn_global_load_async_to_lds_b128)
  asm volatile("s_wait_asynccnt 0" ::: "memory");
#endif
}

__device__ __forceinline__ v16h mk16(v8h lo, v8h hi) {
  return __builtin_shufflevector(lo, hi, 0,1,2,3,4,5,6,7,8,9,10,11,12,13,14,15);
}

// ---------------- weight convert: W[K][N] f32 -> Wt[N][Kp] f16 (transposed, padded) ----
__global__ __launch_bounds__(256) void k_wt(const float* __restrict__ W,
                                            _Float16* __restrict__ Wt,
                                            int K, int N, int Kp) {
  int idx = blockIdx.x * 256 + threadIdx.x;
  if (idx >= N * Kp) return;
  int n = idx / Kp, k = idx % Kp;
  Wt[idx] = (k < K) ? (_Float16)W[k * N + n] : (_Float16)0.f;
}

// ---------------- build edge features ef = [x[src], x[dst], edge_attr, 0pad] (f16) ----
__global__ __launch_bounds__(256) void k_ef(const float* __restrict__ x,
                                            const float* __restrict__ ea,
                                            const int* __restrict__ esrc,
                                            const int* __restrict__ edst,
                                            _Float16* __restrict__ ef) {
  int idx = blockIdx.x * 256 + threadIdx.x;
  if (idx >= M_ * KEF_) return;
  int row = idx / KEF_, c = idx % KEF_;
  int b = row >> 9, e = row & (E_ - 1);
  float v = 0.f;
  if (c < ND_)            v = x[(size_t)(b * N_ + esrc[b * E_ + e]) * ND_ + c];
  else if (c < 2 * ND_)   v = x[(size_t)(b * N_ + edst[b * E_ + e]) * ND_ + (c - ND_)];
  else if (c < 2 * ND_ + ED_) v = ea[(size_t)(b * E_ + e) * ED_ + (c - 2 * ND_)];
  ef[idx] = (_Float16)v;
}

// ---------------- WMMA GEMM, 32x64 tile per wave: C[M][N] = A[M][Kp] @ Bt[N][Kp]^T ------
// act: 0 none, 1 mish.  Optional residual add (f32, stride N). Write f32 and/or f16 out.
__global__ __launch_bounds__(128) void k_gemm(const _Float16* __restrict__ A,
                                              const _Float16* __restrict__ Bt,
                                              const float* __restrict__ bias,
                                              const float* __restrict__ resid,
                                              float* __restrict__ Cf,
                                              _Float16* __restrict__ Ch,
                                              int M, int N, int Kp, int act) {
  const int lane = threadIdx.x & 31;
  const int wave = threadIdx.x >> 5;
  const int ng = N >> 6;                        // 64-column groups
  const int tile = blockIdx.x * 4 + wave;
  const int tm = tile / ng, tg = tile % ng;     // tm indexes 32-row blocks
  if (tm * 32 >= M) return;
  const int half = lane >> 4;
  const int l15  = lane & 15;
  const _Float16* arow0 = A  + (size_t)(tm * 32 + l15) * Kp + 8 * half;
  const _Float16* arow1 = arow0 + (size_t)16 * Kp;
  const _Float16* brow0 = Bt + (size_t)(tg * 64 + l15) * Kp + 8 * half;
  v8f acc[2][4] = {};
  for (int k0 = 0; k0 < Kp; k0 += 32) {
    // batch ALL fragment loads first so they clause and overlap with wmma issue
    v8h a0lo = *(const v8h*)(arow0 + k0);
    v8h a0hi = *(const v8h*)(arow0 + k0 + 16);
    v8h a1lo = *(const v8h*)(arow1 + k0);
    v8h a1hi = *(const v8h*)(arow1 + k0 + 16);
    v8h blo[4], bhi[4];
#pragma unroll
    for (int g = 0; g < 4; ++g) {
      const _Float16* br = brow0 + (size_t)(g * 16) * Kp;
      blo[g] = *(const v8h*)(br + k0);
      bhi[g] = *(const v8h*)(br + k0 + 16);
    }
    v16h a0 = mk16(a0lo, a0hi);
    v16h a1 = mk16(a1lo, a1hi);
#pragma unroll
    for (int g = 0; g < 4; ++g) {
      v16h b = mk16(blo[g], bhi[g]);
      acc[0][g] = __builtin_amdgcn_wmma_f32_16x16x32_f16(false, a0, false, b, (short)0,
                                                         acc[0][g], false, false);
      acc[1][g] = __builtin_amdgcn_wmma_f32_16x16x32_f16(false, a1, false, b, (short)0,
                                                         acc[1][g], false, false);
    }
  }
#pragma unroll
  for (int mi = 0; mi < 2; ++mi) {
#pragma unroll
    for (int g = 0; g < 4; ++g) {
#pragma unroll
      for (int r = 0; r < 8; ++r) {
        int m = tm * 32 + mi * 16 + r + 8 * half;
        int n = tg * 64 + g * 16 + l15;
        float v = acc[mi][g][r] + (bias ? bias[n] : 0.f);
        if (act == 1) v = mishf(v);
        if (resid) v += resid[(size_t)m * N + n];
        if (Cf) Cf[(size_t)m * N + n] = v;
        if (Ch) Ch[(size_t)m * N + n] = (_Float16)v;
      }
    }
  }
}

// ---------------- LayerNorm over HID=128, one wave per row ----------------
__global__ __launch_bounds__(256) void k_ln(const float* __restrict__ h,
                                            const float* __restrict__ g,
                                            const float* __restrict__ bb,
                                            float* __restrict__ hn,
                                            _Float16* __restrict__ hnh, int Mrows) {
  int lane = threadIdx.x & 31;
  int wave = threadIdx.x >> 5;
  int row = blockIdx.x * 8 + wave;
  if (row >= Mrows) return;
  const float* p = h + (size_t)row * HID_;
  float4 xv = *(const float4*)(p + lane * 4);
  float s = xv.x + xv.y + xv.z + xv.w;
#pragma unroll
  for (int off = 16; off; off >>= 1) s += __shfl_xor(s, off, 32);
  float mu = s * (1.f / 128.f);
  float d0 = xv.x - mu, d1 = xv.y - mu, d2 = xv.z - mu, d3 = xv.w - mu;
  float vs = d0 * d0 + d1 * d1 + d2 * d2 + d3 * d3;
#pragma unroll
  for (int off = 16; off; off >>= 1) vs += __shfl_xor(vs, off, 32);
  float inv = rsqrtf(vs * (1.f / 128.f) + 1e-8f);
  float dd[4] = {d0, d1, d2, d3};
#pragma unroll
  for (int j = 0; j < 4; ++j) {
    int c = lane * 4 + j;
    float v = dd[j] * inv * g[c] + bb[c];
    hn[(size_t)row * HID_ + c] = v;
    hnh[(size_t)row * HID_ + c] = (_Float16)v;
  }
}

// ---------------- masked multi-head attention (per batch, head, 32-query tile) ----------
// LDS: 32x512 f16 scores (32KB) + 16KB K-slice (async-staged) reused as V^T + 2KB mask.
__global__ __launch_bounds__(64) void k_attn(const _Float16* __restrict__ qh,
                                             const _Float16* __restrict__ kh,
                                             const _Float16* __restrict__ vh,
                                             _Float16* __restrict__ oh,
                                             const int* __restrict__ esrc,
                                             const int* __restrict__ edst, int masked) {
  __shared__ __align__(16) _Float16 sle[32 * E_];    // score tile      32 KB
  __shared__ __align__(16) _Float16 kvbuf[16 * E_];  // K slice / V^T   16 KB
  __shared__ int ssd[E_];                            // src|dst<<16      2 KB
  __shared__ float red[32];
  const int tid = threadIdx.x;
  const int lane = tid & 31, wave = tid >> 5;
  const int half = lane >> 4, l15 = lane & 15;
  const int qt   = blockIdx.x & 15;
  const int head = (blockIdx.x >> 4) & 7;
  const int b    = blockIdx.x >> 7;
  const int qbase = qt * 32;

  // stage mask data + K head-slice (kvbuf[key*16 + d]) via async global->LDS b128
  for (int t = tid; t < E_; t += 64)
    ssd[t] = (esrc[b * E_ + t] & 0xffff) | (edst[b * E_ + t] << 16);
  for (int t = tid; t < E_ * 2; t += 64) {
    int key = t >> 1, c = (t & 1) * 8;
    copy16_g2l(kh + (size_t)(b * E_ + key) * HID_ + head * 16 + c, kvbuf + key * 16 + c);
  }
  copy_drain();
  __syncthreads();

  // ---- phase 1: scores S = QK^T / 4 (+mask), dh=16 padded to K=32 ----
  v8h zh = {};
  const _Float16* qp = qh + (size_t)(b * E_ + qbase + wave * 16 + l15) * HID_ +
                       head * 16 + 8 * half;
  v16h af = mk16(*(const v8h*)qp, zh);
  for (int kbp = 0; kbp < E_; kbp += 16) {
    v8h klo = *(const v8h*)(kvbuf + (kbp + l15) * 16 + 8 * half);  // ds_load_b128
    v16h bf = mk16(klo, zh);
    v8f sc = {};
    sc = __builtin_amdgcn_wmma_f32_16x16x32_f16(false, af, false, bf, (short)0, sc,
                                                false, false);
    int j = kbp + l15;
    int pj = ssd[j];
    int sj = pj & 0xffff, dj = pj >> 16;
#pragma unroll
    for (int r = 0; r < 8; ++r) {
      int ql = wave * 16 + r + 8 * half;
      int i = qbase + ql;
      float v = sc[r] * 0.25f;
      if (masked) {
        int pi = ssd[i];
        int si = pi & 0xffff, di = pi >> 16;
        bool adj = (si == sj) | (si == dj) | (di == sj) | (di == dj);
        if (!adj) v = -30000.f;
      }
      sle[(size_t)ql * E_ + j] = (_Float16)v;
    }
  }
  __syncthreads();  // also: done reading kvbuf as K

  // ---- phase 2a: re-stage kvbuf as V^T[d][key] (transposed for b128 LDS reads) ----
  for (int key = tid; key < E_; key += 64) {
    const _Float16* vp = vh + (size_t)(b * E_ + key) * HID_ + head * 16;
    v8h lo = *(const v8h*)vp;
    v8h hi = *(const v8h*)(vp + 8);
#pragma unroll
    for (int d = 0; d < 8; ++d) {
      kvbuf[(size_t)d * E_ + key] = lo[d];
      kvbuf[(size_t)(d + 8) * E_ + key] = hi[d];
    }
  }
  // ---- phase 2b: softmax rows (threads 0..31, one row each) ----
  if (tid < 32) {
    _Float16* rowp = sle + (size_t)tid * E_;
    float mx = -1e30f;
    for (int c = 0; c < E_; ++c) mx = fmaxf(mx, (float)rowp[c]);
    float sum = 0.f;
    for (int c = 0; c < E_; ++c) {
      float e = __expf((float)rowp[c] - mx);
      sum += e;
      rowp[c] = (_Float16)e;  // unnormalized probs; divide in epilogue
    }
    red[tid] = sum;
  }
  __syncthreads();

  // ---- phase 3: O = A @ V (K = keys, full 32 per step; all operands from LDS) ----
  v8f oacc = {};
  const _Float16* sp = sle + (size_t)(wave * 16 + l15) * E_ + 8 * half;
  const _Float16* vrow = kvbuf + (size_t)l15 * E_ + 8 * half;   // V^T row for dim l15
  for (int kb = 0; kb < E_; kb += 32) {
    v16h aa = mk16(*(const v8h*)(sp + kb), *(const v8h*)(sp + kb + 16));
    v16h bv = mk16(*(const v8h*)(vrow + kb), *(const v8h*)(vrow + kb + 16));
    oacc = __builtin_amdgcn_wmma_f32_16x16x32_f16(false, aa, false, bv, (short)0, oacc,
                                                  false, false);
  }
#pragma unroll
  for (int r = 0; r < 8; ++r) {
    int ql = wave * 16 + r + 8 * half;
    float v = oacc[r] / red[ql];
    int grow = b * E_ + qbase + ql;
    oh[(size_t)grow * HID_ + head * 16 + l15] = (_Float16)v;
  }
}

// ---------------- PMA pooling + output head (one block per batch, deterministic) -------
__global__ __launch_bounds__(128) void k_pma_out(const _Float16* __restrict__ kh,
                                                 const _Float16* __restrict__ vh,
                                                 const float* __restrict__ seed,
                                                 const float* __restrict__ Wq,
                                                 const float* __restrict__ bq,
                                                 const float* __restrict__ Wo,
                                                 const float* __restrict__ bo,
                                                 const float* __restrict__ oW1,
                                                 const float* __restrict__ ob1,
                                                 const float* __restrict__ oW2,
                                                 const float* __restrict__ ob2,
                                                 float* __restrict__ out) {
  __shared__ float qv[HID_], ov[HID_], pool[HID_], m1[HID_], sred[128];
  __shared__ float pvs[128][16];
  const int b = blockIdx.x, t = threadIdx.x;

  float acc = bq[t];
  for (int k = 0; k < HID_; ++k) acc += seed[k] * Wq[k * HID_ + t];
  qv[t] = acc;
  __syncthreads();

  for (int hh = 0; hh < H_; ++hh) {
    const _Float16* kbp = kh + (size_t)(b * E_) * HID_ + hh * 16;
    const _Float16* vbp = vh + (size_t)(b * E_) * HID_ + hh * 16;
    float s4[4], lmax = -1e30f;
#pragma unroll
    for (int jj = 0; jj < 4; ++jj) {
      int j = t + jj * 128;
      const _Float16* kp = kbp + (size_t)j * HID_;
      float a2 = 0.f;
#pragma unroll
      for (int d = 0; d < 16; ++d) a2 += qv[hh * 16 + d] * (float)kp[d];
      s4[jj] = a2 * 0.25f;
      lmax = fmaxf(lmax, s4[jj]);
    }
    sred[t] = lmax; __syncthreads();
    for (int off = 64; off; off >>= 1) { if (t < off) sred[t] = fmaxf(sred[t], sred[t + off]); __syncthreads(); }
    float mx = sred[0]; __syncthreads();

    float pvv[16];
#pragma unroll
    for (int d = 0; d < 16; ++d) pvv[d] = 0.f;
    float lsum = 0.f;
#pragma unroll
    for (int jj = 0; jj < 4; ++jj) {
      int j = t + jj * 128;
      float e = __expf(s4[jj] - mx);
      lsum += e;
      const _Float16* vp = vbp + (size_t)j * HID_;
#pragma unroll
      for (int d = 0; d < 16; ++d) pvv[d] += e * (float)vp[d];
    }
    sred[t] = lsum; __syncthreads();
    for (int off = 64; off; off >>= 1) { if (t < off) sred[t] += sred[t + off]; __syncthreads(); }
    float ssum = sred[0]; __syncthreads();
#pragma unroll
    for (int d = 0; d < 16; ++d) pvs[t][d] = pvv[d];
    __syncthreads();
    if (t < 16) {
      float oa = 0.f;
      for (int i = 0; i < 128; ++i) oa += pvs[i][t];
      ov[hh * 16 + t] = oa / ssum;
    }
    __syncthreads();
  }

  float pacc = seed[t] + bo[t];
  for (int k = 0; k < HID_; ++k) pacc += ov[k] * Wo[k * HID_ + t];
  pool[t] = pacc;
  __syncthreads();
  float macc = ob1[t];
  for (int k = 0; k < HID_; ++k) macc += pool[k] * oW1[k * HID_ + t];
  m1[t] = mishf(macc);
  __syncthreads();
  sred[t] = m1[t] * oW2[t];
  __syncthreads();
  for (int off = 64; off; off >>= 1) { if (t < off) sred[t] += sred[t + off]; __syncthreads(); }
  if (t == 0) out[b] = sred[0] + ob2[0];
}

// ================= host =================
extern "C" void kernel_launch(void* const* d_in, const int* in_sizes, int n_in,
                              void* d_out, int out_size, void* d_ws, size_t ws_size,
                              hipStream_t stream) {
  (void)in_sizes; (void)n_in; (void)out_size; (void)ws_size;
  const float* x        = (const float*)d_in[0];
  const float* eattr    = (const float*)d_in[1];
  const float* mlp_W1   = (const float*)d_in[2];
  const float* mlp_b1   = (const float*)d_in[3];
  const float* mlp_W2   = (const float*)d_in[4];
  const float* mlp_b2   = (const float*)d_in[5];
  const float* eln_g    = (const float*)d_in[6];
  const float* eln_b    = (const float*)d_in[7];
  const float* eWq      = (const float*)d_in[8];
  const float* ebq      = (const float*)d_in[9];
  const float* eWk      = (const float*)d_in[10];
  const float* ebk      = (const float*)d_in[11];
  const float* eWv      = (const float*)d_in[12];
  const float* ebv      = (const float*)d_in[13];
  const float* eWo      = (const float*)d_in[14];
  const float* ebo      = (const float*)d_in[15];
  const float* dln_g    = (const float*)d_in[16];
  const float* dln_b    = (const float*)d_in[17];
  const float* pseed    = (const float*)d_in[18];
  const float* pWq      = (const float*)d_in[19];
  const float* pbq      = (const float*)d_in[20];
  const float* pWk      = (const float*)d_in[21];
  const float* pbk      = (const float*)d_in[22];
  const float* pWv      = (const float*)d_in[23];
  const float* pbv      = (const float*)d_in[24];
  const float* pWo      = (const float*)d_in[25];
  const float* pbo      = (const float*)d_in[26];
  const float* oW1      = (const float*)d_in[27];
  const float* ob1      = (const float*)d_in[28];
  const float* oW2      = (const float*)d_in[29];
  const float* ob2      = (const float*)d_in[30];
  const int*   esrc     = (const int*)d_in[31];
  const int*   edst     = (const int*)d_in[32];
  float* out = (float*)d_out;

  char* base = (char*)d_ws;
  size_t off = 0;
  auto take = [&](size_t bytes) -> void* {
    void* p = base + off;
    off += (bytes + 255) & ~(size_t)255;
    return p;
  };
  _Float16* W1t = (_Float16*)take(HID_ * KEF_ * 2);
  _Float16* W2t = (_Float16*)take(HID_ * HID_ * 2);
  _Float16 *Wqt[3], *Wkt[3], *Wvt[3], *Wot[3];
  for (int i = 0; i < 3; ++i) {
    Wqt[i] = (_Float16*)take(HID_ * HID_ * 2);
    Wkt[i] = (_Float16*)take(HID_ * HID_ * 2);
    Wvt[i] = (_Float16*)take(HID_ * HID_ * 2);
    Wot[i] = (_Float16*)take(HID_ * HID_ * 2);
  }
  _Float16* pKt = (_Float16*)take(HID_ * HID_ * 2);
  _Float16* pVt = (_Float16*)take(HID_ * HID_ * 2);
  float*    hbuf = (float*)take((size_t)M_ * HID_ * 4);
  float*    hn   = (float*)take((size_t)M_ * HID_ * 4);
  _Float16* hnh  = (_Float16*)take((size_t)M_ * HID_ * 2);
  _Float16* qh   = (_Float16*)take((size_t)M_ * HID_ * 2);
  _Float16* kh   = (_Float16*)take((size_t)M_ * HID_ * 2);
  _Float16* vh   = (_Float16*)take((size_t)M_ * HID_ * 2);
  _Float16* ohb  = (_Float16*)take((size_t)M_ * HID_ * 2);
  _Float16* efh  = ohb;   // alias: ef (6MB) used before o (8MB)
  _Float16* mlph = qh;    // alias: mlp hidden used before q

  auto wt = [&](const float* W, _Float16* Wt, int K, int N, int Kp) {
    int n = N * Kp;
    hipLaunchKernelGGL(k_wt, dim3((n + 255) / 256), dim3(256), 0, stream, W, Wt, K, N, Kp);
  };
  auto gemm = [&](const _Float16* A, const _Float16* Bt, const float* bias,
                  const float* resid, float* Cf, _Float16* Ch, int M, int N, int Kp,
                  int act) {
    int tiles = (M / 32) * (N >> 6);   // 32x64 tile per wave, 4 waves per block
    hipLaunchKernelGGL(k_gemm, dim3(tiles / 4), dim3(128), 0, stream, A, Bt, bias, resid,
                       Cf, Ch, M, N, Kp, act);
  };

  // weights -> f16 transposed
  wt(mlp_W1, W1t, 2 * ND_ + ED_, HID_, KEF_);
  wt(mlp_W2, W2t, HID_, HID_, HID_);
  for (int i = 0; i < 3; ++i) {
    wt(eWq + (size_t)i * HID_ * HID_, Wqt[i], HID_, HID_, HID_);
    wt(eWk + (size_t)i * HID_ * HID_, Wkt[i], HID_, HID_, HID_);
    wt(eWv + (size_t)i * HID_ * HID_, Wvt[i], HID_, HID_, HID_);
    wt(eWo + (size_t)i * HID_ * HID_, Wot[i], HID_, HID_, HID_);
  }
  wt(pWk, pKt, HID_, HID_, HID_);
  wt(pWv, pVt, HID_, HID_, HID_);

  // edge features + edge MLP
  hipLaunchKernelGGL(k_ef, dim3((M_ * KEF_ + 255) / 256), dim3(256), 0, stream, x, eattr,
                     esrc, edst, efh);
  gemm(efh, W1t, mlp_b1, nullptr, nullptr, mlph, M_, HID_, KEF_, 1);
  gemm(mlph, W2t, mlp_b2, nullptr, hbuf, nullptr, M_, HID_, HID_, 0);

  // 3 encoder layers
  for (int i = 0; i < 3; ++i) {
    hipLaunchKernelGGL(k_ln, dim3(M_ / 8), dim3(256), 0, stream, hbuf,
                       eln_g + i * HID_, eln_b + i * HID_, hn, hnh, M_);
    gemm(hnh, Wqt[i], ebq + i * HID_, nullptr, nullptr, qh, M_, HID_, HID_, 0);
    gemm(hnh, Wkt[i], ebk + i * HID_, nullptr, nullptr, kh, M_, HID_, HID_, 0);
    gemm(hnh, Wvt[i], ebv + i * HID_, nullptr, nullptr, vh, M_, HID_, HID_, 0);
    hipLaunchKernelGGL(k_attn, dim3(B_ * H_ * (E_ / 32)), dim3(64), 0, stream, qh, kh,
                       vh, ohb, esrc, edst, (i < 2) ? 1 : 0);
    gemm(ohb, Wot[i], ebo + i * HID_, hn, hbuf, nullptr, M_, HID_, HID_, 0);
  }

  // decoder LN + PMA K/V projections + pooling head
  hipLaunchKernelGGL(k_ln, dim3(M_ / 8), dim3(256), 0, stream, hbuf, dln_g, dln_b, hn,
                     hnh, M_);
  gemm(hnh, pKt, pbk, nullptr, nullptr, kh, M_, HID_, HID_, 0);
  gemm(hnh, pVt, pbv, nullptr, nullptr, vh, M_, HID_, HID_, 0);
  hipLaunchKernelGGL(k_pma_out, dim3(B_), dim3(128), 0, stream, kh, vh, pseed, pWq, pbq,
                     pWo, pbo, oW1, ob1, oW2, ob2, out);
}
